// GATLayer_49881750175984
// MI455X (gfx1250) — compile-verified, matrix operand order checked by reading, not run on hardware
//
#include <hip/hip_runtime.h>
#include <hip/hip_bf16.h>
#include <math.h>

// GAT layer, fused flash-style for MI455X (gfx1250, wave32, WMMA):
//   K0: Wt[n][k]  = bf16(W[k][n])            (K-major B operand for K1)
//   K1: Wh = X @ W  -> Wh f32 + WhbT[n][j]   (8x v_wmma_f32_16x16x32_bf16/wave)
//   K2: s_src/s_dst = Wh @ a halves
//   K3: one pass over A (256 MB = roofline floor): bit-compress adjacency
//       (-> 8 MB) + online softmax row stats via wave32 __ballot + prefetch
//   K4: partial[chunk] = softmax-probs @ Wh; B slabs staged into LDS with
//       double-buffered global_load_async_to_lds_b128 (+ s_wait_asynccnt),
//       fragments read as ds_load_b128 from a bank-conflict-padded layout;
//       p_ij regenerated from stats; sdst gathered via ds_bpermute/__shfl
//   K5: out = ELU(sum of 4 partials)

#define N_NODES 8192
#define F_IN    512
#define F_OUT   128
#define NWORDS  (N_NODES / 32)
#define NEG_BIG (-9.0e15f)
#define JSPLIT  4
#define JCHUNK  (N_NODES / JSPLIT)
#define BSTRIDE 40   // bf16 elems per LDS B-slab row: 32 data + 8 pad (80 B)

typedef __attribute__((ext_vector_type(16))) __bf16 v16bf;
typedef __attribute__((ext_vector_type(8)))  __bf16 v8bf;
typedef __attribute__((ext_vector_type(8)))  float  v8f;

// K index of the low element held in 16-bit A/B fragment VGPR v, lane-half hf.
// (ISA 7.12.2: 16-bit A-matrix 16x32; B mirrors with lane = N.)
__device__ __forceinline__ int frag_klo(int v, int hf) {
    return ((v >> 2) << 4) + (hf << 3) + ((v & 3) << 1);
}

// ------------------------------------------- K0: Wt[n][k] = bf16(W[k][n])
__global__ void gat_wtrans(const float* __restrict__ W,
                           unsigned short* __restrict__ Wt_u)
{
    __bf16* Wt = (__bf16*)Wt_u;
    const int idx = blockIdx.x * blockDim.x + threadIdx.x;   // < 128*512
    const int n = idx >> 9;          // /512
    const int k = idx & 511;
    Wt[(size_t)n * F_IN + k] = (__bf16)W[(size_t)k * F_OUT + n];
}

// ---------------------------------- K1: Wh = X @ W (one wave -> 16x128 stripe)
__global__ void gat_wh_wmma(const float* __restrict__ X,
                            const unsigned short* __restrict__ Wt_u,
                            float* __restrict__ Wh,
                            unsigned short* __restrict__ WhbT_u)
{
    const __bf16* Wt = (const __bf16*)Wt_u;
    __bf16* WhbT = (__bf16*)WhbT_u;
    const int lane = threadIdx.x;          // one wave32 per block
    const int i0 = blockIdx.x * 16;        // output row stripe
    const int MN = lane & 15;              // row for A-frag, col for B-frag
    const int hf = lane >> 4;

    v8f c[8];
#pragma unroll
    for (int t = 0; t < 8; ++t) c[t] = (v8f){};

    for (int k0 = 0; k0 < F_IN; k0 += 32) {
        // A fragment: X[i0+MN][k0 + g*16 + hf*8 + 0..7], converted to bf16.
        v16bf a;
#pragma unroll
        for (int g = 0; g < 2; ++g) {
            const float* src = &X[(size_t)(i0 + MN) * F_IN + k0 + g * 16 + hf * 8];
            float4 f0 = *(const float4*)src;
            float4 f1 = *(const float4*)(src + 4);
            a[g * 8 + 0] = (__bf16)f0.x; a[g * 8 + 1] = (__bf16)f0.y;
            a[g * 8 + 2] = (__bf16)f0.z; a[g * 8 + 3] = (__bf16)f0.w;
            a[g * 8 + 4] = (__bf16)f1.x; a[g * 8 + 5] = (__bf16)f1.y;
            a[g * 8 + 6] = (__bf16)f1.z; a[g * 8 + 7] = (__bf16)f1.w;
        }
#pragma unroll
        for (int t = 0; t < 8; ++t) {
            // B fragment: Wt[t*16+MN][k0 + g*16 + hf*8 + 0..7] (two 16B loads)
            v16bf b;
#pragma unroll
            for (int g = 0; g < 2; ++g) {
                v8bf chunk = *(const v8bf*)&Wt[(size_t)(t * 16 + MN) * F_IN +
                                               k0 + g * 16 + hf * 8];
#pragma unroll
                for (int q = 0; q < 8; ++q) b[g * 8 + q] = chunk[q];
            }
            c[t] = __builtin_amdgcn_wmma_f32_16x16x32_bf16(false, a, false, b,
                                                           (short)0, c[t],
                                                           false, false);
        }
    }
#pragma unroll
    for (int t = 0; t < 8; ++t) {
        v8bf pack;
#pragma unroll
        for (int r = 0; r < 8; ++r) {
            const int row = i0 + r + hf * 8;   // C layout: VGPR r -> M = r + 8*half
            Wh[(size_t)row * F_OUT + t * 16 + MN] = c[t][r];
            pack[r] = (__bf16)c[t][r];
        }
        // rows i0+hf*8 .. +7 are contiguous along WhbT's j axis: one 16B store
        *(v8bf*)&WhbT[(size_t)(t * 16 + MN) * N_NODES + i0 + hf * 8] = pack;
    }
}

// ------------------------------------------- K2: s_src/s_dst = Wh @ a halves
__global__ void gat_scores(const float* __restrict__ Wh,
                           const float* __restrict__ avec,
                           float* __restrict__ ssrc,
                           float* __restrict__ sdst)
{
    __shared__ float r0[F_OUT], r1[F_OUT];
    const int i = blockIdx.x;
    const int t = threadIdx.x;             // 0..127
    const float w = Wh[(size_t)i * F_OUT + t];
    r0[t] = w * avec[t];
    r1[t] = w * avec[F_OUT + t];
    __syncthreads();
    for (int s = F_OUT / 2; s > 0; s >>= 1) {
        if (t < s) { r0[t] += r0[t + s]; r1[t] += r1[t + s]; }
        __syncthreads();
    }
    if (t == 0) { ssrc[i] = r0[0]; sdst[i] = r1[0]; }
}

// ----------- K3: compress A to bits + online softmax row stats (single pass)
__global__ void gat_mask_stats(const int* __restrict__ A,
                               const float* __restrict__ ssrc,
                               const float* __restrict__ sdst,
                               unsigned* __restrict__ mask,
                               float* __restrict__ rowm,
                               float* __restrict__ rowinv)
{
    __shared__ float sm[256], ss[256];
    __shared__ int   sc[256];
    const int i    = blockIdx.x;
    const int t    = threadIdx.x;          // 256 threads = 8 waves
    const int lane = t & 31;
    const int wv   = t >> 5;
    const float si = ssrc[i];
    const size_t rowbase = (size_t)i * N_NODES;

    float m = NEG_BIG, s = 0.0f;
    int cnt = 0;
    for (int w = wv; w < NWORDS; w += 8) {
        const int j = w * 32 + lane;       // coalesced: wave reads one 128B line
        // speculative prefetch of this wave's next line (dropped if OOB)
        __builtin_prefetch(&A[rowbase + (size_t)(w + 8) * 32 + lane], 0, 0);
        const bool on = A[rowbase + j] > 0;
        const unsigned bits = (unsigned)__ballot(on);   // wave32 mask word
        if (lane == 0) mask[(size_t)i * NWORDS + w] = bits;
        if (on) {
            float e = si + sdst[j];
            e = e > 0.0f ? e : 0.0f;       // relu
            const float mn = fmaxf(m, e);  // online softmax merge
            s = s * __expf(m - mn) + __expf(e - mn);
            m = mn;
            ++cnt;
        }
    }
    sm[t] = m; ss[t] = s; sc[t] = cnt;
    __syncthreads();
    for (int r = 128; r > 0; r >>= 1) {
        if (t < r) {
            const float m2 = sm[t + r], s2 = ss[t + r];
            const float mn = fmaxf(sm[t], m2);
            ss[t] = ss[t] * __expf(sm[t] - mn) + s2 * __expf(m2 - mn);
            sm[t] = mn;
            sc[t] += sc[t + r];
        }
        __syncthreads();
    }
    if (t == 0) {
        rowm[i] = sm[0];
        // empty row: softmax over all -9e15 -> uniform 1/N. Negative invZ flags it.
        rowinv[i] = (sc[0] == 0) ? (-1.0f / (float)N_NODES) : (1.0f / ss[0]);
    }
}

// ---- K4: partial = P @ Wh over one j-chunk (one wave -> 16x128 f32 stripe)
__global__ void gat_attn_wmma(const unsigned* __restrict__ mask,
                              const float* __restrict__ ssrc,
                              const float* __restrict__ sdst,
                              const float* __restrict__ rowm,
                              const float* __restrict__ rowinv,
                              const unsigned short* __restrict__ WhbT_u,
                              float* __restrict__ partial)
{
    // Double-buffered LDS stage for the 32(K) x 128(N) bf16 B-slab.
    // K-major rows padded to 80 B so the 16B fragment reads of lanes 0..15
    // start on disjoint 4-bank groups (stride = 20 banks mod 64).
    __shared__ __align__(16) __bf16 sB[2][F_OUT * BSTRIDE];

    const __bf16* WhbT = (const __bf16*)WhbT_u;
    const int lane  = threadIdx.x;         // one wave32 per block
    const int i0    = blockIdx.x * 16;
    const int chunk = blockIdx.y;          // j in [chunk*JCHUNK, +JCHUNK)
    const int MN = lane & 15;
    const int hf = lane >> 4;

    // Per-lane row constants (A-fragment row M = MN)
    const float si = ssrc[i0 + MN];
    const float mi = rowm[i0 + MN];
    const float zi = rowinv[i0 + MN];
    const bool  empty = zi < 0.0f;
    const float uni   = -zi;
    const unsigned* mrow = mask + (size_t)(i0 + MN) * NWORDS;

    // Coalesced async fill: 512 x 16B transfers, id = it*32 + lane.
    // n = id/4 (slab row), kq = (id%4)*8 (k offset in elements).
    auto fill = [&](int bufsel, int j0) {
        const unsigned lbase = (unsigned)(size_t)&sB[bufsel][0];
#pragma unroll
        for (int it = 0; it < 16; ++it) {
            const int id = it * 32 + lane;
            const int n  = id >> 2;
            const int kq = (id & 3) << 3;
            const unsigned laddr = lbase + (unsigned)(n * BSTRIDE + kq) * 2u;
            const unsigned long long gaddr =
                (unsigned long long)(size_t)&WhbT[(size_t)n * N_NODES + j0 + kq];
            asm volatile("global_load_async_to_lds_b128 %0, %1, off"
                         :: "v"(laddr), "v"(gaddr) : "memory");
        }
    };

    v8f c[8];
#pragma unroll
    for (int t = 0; t < 8; ++t) c[t] = (v8f){};

    const int jbeg = chunk * JCHUNK;
    const int jend = jbeg + JCHUNK;
    int b = 0;
    fill(0, jbeg);
    for (int j0 = jbeg; j0 < jend; j0 += 32) {
        const int nxt = j0 + 32;
        if (nxt < jend) {
            fill(b ^ 1, nxt);
            // current buffer's 16 fills are done once <=16 remain outstanding
            asm volatile("s_wait_asynccnt 0x10" ::: "memory");
        } else {
            asm volatile("s_wait_asynccnt 0x0" ::: "memory");
        }

        const unsigned mw = mrow[j0 >> 5];        // all 32 K-bits for this row
        __builtin_prefetch(&mrow[(j0 >> 5) + 1], 0, 0);
        const float sd = sdst[j0 + lane];         // fragment values gathered
                                                  // via ds_bpermute (__shfl)
        // A fragment: softmax probabilities, 16 exp/lane
        v16bf a;
#pragma unroll
        for (int v = 0; v < 8; ++v) {
            const int kl = frag_klo(v, hf);
#pragma unroll
            for (int u = 0; u < 2; ++u) {
                const int k = kl + u;
                const float sdk = __shfl(sd, k, 32);
                float p;
                if (empty) {
                    p = uni;
                } else if ((mw >> k) & 1u) {
                    float e = si + sdk;
                    e = e > 0.0f ? e : 0.0f;
                    p = __expf(e - mi) * zi;
                } else {
                    p = 0.0f;
                }
                a[2 * v + u] = (__bf16)p;
            }
        }
        // 8 N-tiles reuse the A fragment; B fragments via ds_load_b128
        const __bf16* bufp = &sB[b][0];
#pragma unroll
        for (int t = 0; t < 8; ++t) {
            v16bf bb;
#pragma unroll
            for (int g = 0; g < 2; ++g) {
                const v8bf ch = *(const v8bf*)&bufp[(t * 16 + MN) * BSTRIDE +
                                                    g * 16 + hf * 8];
#pragma unroll
                for (int q = 0; q < 8; ++q) bb[g * 8 + q] = ch[q];
            }
            c[t] = __builtin_amdgcn_wmma_f32_16x16x32_bf16(false, a, false, bb,
                                                           (short)0, c[t],
                                                           false, false);
        }
        b ^= 1;
    }
    float* pout = partial + (size_t)chunk * N_NODES * F_OUT;
#pragma unroll
    for (int t = 0; t < 8; ++t)
#pragma unroll
        for (int r = 0; r < 8; ++r)
            pout[(size_t)(i0 + r + hf * 8) * F_OUT + t * 16 + MN] = c[t][r];
}

// -------------------------------------- K5: out = ELU(sum of JSPLIT partials)
__global__ void gat_finish(const float* __restrict__ partial,
                           float* __restrict__ out)
{
    const size_t idx = (size_t)blockIdx.x * blockDim.x + threadIdx.x;
    const size_t M = (size_t)N_NODES * F_OUT;
    float v = 0.0f;
#pragma unroll
    for (int s = 0; s < JSPLIT; ++s) v += partial[s * M + idx];
    out[idx] = v > 0.0f ? v : (__expf(v) - 1.0f);   // ELU (alpha=1)
}

// --------------------------------------------------------------------- launcher
extern "C" void kernel_launch(void* const* d_in, const int* in_sizes, int n_in,
                              void* d_out, int out_size, void* d_ws, size_t ws_size,
                              hipStream_t stream)
{
    const float* X    = (const float*)d_in[0];   // [8192,512]
    const int*   A    = (const int*)  d_in[1];   // [8192,8192]
    const float* W    = (const float*)d_in[2];   // [512,128]
    const float* avec = (const float*)d_in[3];   // [256,1]

    // Workspace carve-up (~30.5 MB, every byte written before read)
    char* ws = (char*)d_ws;
    float* Wh            = (float*)ws;          ws += (size_t)N_NODES * F_OUT * sizeof(float);
    unsigned short* WhbT = (unsigned short*)ws; ws += (size_t)N_NODES * F_OUT * sizeof(unsigned short);
    unsigned short* Wt   = (unsigned short*)ws; ws += (size_t)F_IN * F_OUT * sizeof(unsigned short);
    float* ssrc          = (float*)ws;          ws += (size_t)N_NODES * sizeof(float);
    float* sdst          = (float*)ws;          ws += (size_t)N_NODES * sizeof(float);
    float* rowm          = (float*)ws;          ws += (size_t)N_NODES * sizeof(float);
    float* rowinv        = (float*)ws;          ws += (size_t)N_NODES * sizeof(float);
    unsigned* mask       = (unsigned*)ws;       ws += (size_t)N_NODES * NWORDS * sizeof(unsigned);
    float* partial       = (float*)ws;          ws += (size_t)JSPLIT * N_NODES * F_OUT * sizeof(float);

    gat_wtrans    <<<(F_IN * F_OUT) / 256, 256, 0, stream>>>(W, Wt);
    gat_wh_wmma   <<<N_NODES / 16, 32, 0, stream>>>(X, Wt, Wh, WhbT);
    gat_scores    <<<N_NODES, 128, 0, stream>>>(Wh, avec, ssrc, sdst);
    gat_mask_stats<<<N_NODES, 256, 0, stream>>>(A, ssrc, sdst, mask, rowm, rowinv);
    dim3 agrid(N_NODES / 16, JSPLIT);            // 2048 waves
    gat_attn_wmma <<<agrid, 32, 0, stream>>>(mask, ssrc, sdst, rowm, rowinv,
                                             WhbT, partial);
    gat_finish    <<<(N_NODES * F_OUT) / 256, 256, 0, stream>>>(partial, (float*)d_out);
}
